// BondLevel_7275674599785
// MI455X (gfx1250) — matrix operand order repaired and checked by reading, not compiled, not run on hardware
//
#include <hip/hip_runtime.h>
#include <hip/hip_bf16.h>
#include <math.h>

#define D 64
#define HFEAT 128

typedef __attribute__((ext_vector_type(16))) __bf16 v16bf;
typedef __attribute__((ext_vector_type(8)))  __bf16 v8bf;
typedef __attribute__((ext_vector_type(8)))  float  v8f;

#define WMMA_BF16(a, b, c) \
  __builtin_amdgcn_wmma_f32_16x16x32_bf16(false, (a), false, (b), (short)0, (c), false, false)

static __device__ __forceinline__ float silu_f(float v) {
  return v / (1.0f + __expf(-v));
}

// Build an A-fragment (16 bf16 per lane) from two contiguous 8-float runs.
// Element e<8  -> K = 32k + 8h + e        (run0)
// Element e>=8 -> K = 32k + 16 + 8h + e-8 (run1)
static __device__ __forceinline__ v16bf make_a(float4 a0, float4 a1, float4 b0, float4 b1) {
  v16bf r;
  r[0]  = (__bf16)a0.x; r[1]  = (__bf16)a0.y; r[2]  = (__bf16)a0.z; r[3]  = (__bf16)a0.w;
  r[4]  = (__bf16)a1.x; r[5]  = (__bf16)a1.y; r[6]  = (__bf16)a1.z; r[7]  = (__bf16)a1.w;
  r[8]  = (__bf16)b0.x; r[9]  = (__bf16)b0.y; r[10] = (__bf16)b0.z; r[11] = (__bf16)b0.w;
  r[12] = (__bf16)b1.x; r[13] = (__bf16)b1.y; r[14] = (__bf16)b1.z; r[15] = (__bf16)b1.w;
  return r;
}

static __device__ __forceinline__ v16bf gather_a(const float* p0, const float* p1) {
  const float4* q0 = (const float4*)p0;
  const float4* q1 = (const float4*)p1;
  return make_a(q0[0], q0[1], q1[0], q1[1]);
}

static __device__ __forceinline__ v16bf gather_a_scaled(const float* p0, float s0,
                                                        const float* p1, float s1) {
  float4 a0 = ((const float4*)p0)[0], a1 = ((const float4*)p0)[1];
  float4 b0 = ((const float4*)p1)[0], b1 = ((const float4*)p1)[1];
  a0.x *= s0; a0.y *= s0; a0.z *= s0; a0.w *= s0;
  a1.x *= s0; a1.y *= s0; a1.z *= s0; a1.w *= s0;
  b0.x *= s1; b0.y *= s1; b0.z *= s1; b0.w *= s1;
  b1.x *= s1; b1.y *= s1; b1.z *= s1; b1.w *= s1;
  return make_a(a0, a1, b0, b1);
}

// Pre-packed B fragments: per (n_tile, k_step) block of 32 lanes x 16 bf16.
// Lane layout: n = nt*16 + lane%16, element e -> K = 32k + 8*(lane/16) + e%8 + 16*(e/8).
static __device__ __forceinline__ v16bf load_b(const __bf16* wp, int nt, int k, int nK, int lane) {
  const v16bf* p = (const v16bf*)(wp + (((size_t)(nt * nK + k)) * 32 + (size_t)lane) * 16);
  return *p;
}

// Load second-GEMM A fragment from an LDS row (row-major, stride 72 bf16).
static __device__ __forceinline__ v16bf lds_a(const __bf16* row, int k, int h) {
  const __bf16* r0 = row + 32 * k + 8 * h;
  v8bf u = *(const v8bf*)r0;
  v8bf w = *(const v8bf*)(r0 + 16);
  return __builtin_shufflevector(u, w, 0, 1, 2, 3, 4, 5, 6, 7, 8, 9, 10, 11, 12, 13, 14, 15);
}

// ---------------- weight packing: f32 [K x 64] -> fragment-ready bf16 --------
__global__ void k_pack(const float* __restrict__ W, __bf16* __restrict__ out, int K) {
  int idx = blockIdx.x * blockDim.x + threadIdx.x;
  int tot = K * D;
  if (idx >= tot) return;
  int e    = idx & 15;
  int lane = (idx >> 4) & 31;
  int rest = idx >> 9;
  int nK   = K >> 5;
  int k    = rest % nK;
  int nt   = rest / nK;
  int h    = lane >> 4;
  int n    = nt * 16 + (lane & 15);
  int Ki   = 32 * k + 8 * h + (e & 7) + ((e >> 3) << 4);
  out[idx] = (__bf16)W[(size_t)Ki * D + n];
}

// ---------------- bond classification (cosine-sim thresholds) ----------------
__global__ void k_classify(const float* __restrict__ af, const int* __restrict__ erow,
                           const int* __restrict__ ecol, int* __restrict__ bt, int E) {
  int e = blockIdx.x * blockDim.x + threadIdx.x;
  if (e >= E) return;
  const float4* hi = (const float4*)(af + (size_t)erow[e] * HFEAT);
  const float4* hj = (const float4*)(af + (size_t)ecol[e] * HFEAT);
  float dot = 0.f, ni = 0.f, nj = 0.f;
#pragma unroll 8
  for (int i = 0; i < HFEAT / 4; i++) {
    float4 a = hi[i], b = hj[i];
    dot += a.x * b.x + a.y * b.y + a.z * b.z + a.w * b.w;
    ni  += a.x * a.x + a.y * a.y + a.z * a.z + a.w * a.w;
    nj  += b.x * b.x + b.y * b.y + b.z * b.z + b.w * b.w;
  }
  ni = fmaxf(sqrtf(ni), 1e-8f);
  nj = fmaxf(sqrtf(nj), 1e-8f);
  float sim = dot / (ni * nj);
  int t = 0;
  if (sim > 0.8f) t = 1;
  if (sim > 0.9f) t = 2;
  if (sim < 0.3f) t = 3;
  bt[e] = t;
}

// ---------------- bond init MLP: x = silu([hi|hj] W1 + b1) W2 + b2 -----------
__global__ void k_bond_init(const float* __restrict__ af, const int* __restrict__ erow,
                            const int* __restrict__ ecol, const __bf16* __restrict__ w1p,
                            const float* __restrict__ b1, const __bf16* __restrict__ w2p,
                            const float* __restrict__ b2, float* __restrict__ x, int E) {
  __shared__ __align__(16) __bf16 sh[16][72];
  int lane = threadIdx.x, m = lane & 15, h = lane >> 4;
  int tile = blockIdx.x;
  int e = tile * 16 + m;
  if (e >= E) e = E - 1;
  const float* hi = af + (size_t)erow[e] * HFEAT;
  const float* hj = af + (size_t)ecol[e] * HFEAT;

  v8f acc[4];
#pragma unroll
  for (int nt = 0; nt < 4; nt++) acc[nt] = (v8f){0, 0, 0, 0, 0, 0, 0, 0};
#pragma unroll
  for (int k = 0; k < 8; k++) {   // K = 256
    int K0 = 32 * k + 8 * h, K1 = K0 + 16;
    const float* p0 = (K0 < HFEAT) ? (hi + K0) : (hj + (K0 - HFEAT));
    const float* p1 = (K1 < HFEAT) ? (hi + K1) : (hj + (K1 - HFEAT));
    v16bf a = gather_a(p0, p1);
#pragma unroll
    for (int nt = 0; nt < 4; nt++)
      acc[nt] = WMMA_BF16(a, load_b(w1p, nt, k, 8, lane), acc[nt]);
  }
#pragma unroll
  for (int nt = 0; nt < 4; nt++) {
    int n = nt * 16 + m;
    float bb = b1[n];
#pragma unroll
    for (int i = 0; i < 8; i++) sh[i + 8 * h][n] = (__bf16)silu_f(acc[nt][i] + bb);
  }
  __syncthreads();

  v8f o[4];
#pragma unroll
  for (int nt = 0; nt < 4; nt++) o[nt] = (v8f){0, 0, 0, 0, 0, 0, 0, 0};
#pragma unroll
  for (int k = 0; k < 2; k++) {   // K = 64
    v16bf a = lds_a(&sh[m][0], k, h);
#pragma unroll
    for (int nt = 0; nt < 4; nt++)
      o[nt] = WMMA_BF16(a, load_b(w2p, nt, k, 2, lane), o[nt]);
  }
#pragma unroll
  for (int nt = 0; nt < 4; nt++) {
    int n = nt * 16 + m;
    float bb = b2[n];
#pragma unroll
    for (int i = 0; i < 8; i++) {
      int er = tile * 16 + i + 8 * h;
      if (er < E) x[(size_t)er * D + n] = o[nt][i] + bb;
    }
  }
}

// ---------------- bond-edge degree --------------------------------------------
__global__ void k_deg(const int* __restrict__ bdst, float* __restrict__ deg, int BE) {
  int i = blockIdx.x * blockDim.x + threadIdx.x;
  if (i < BE) atomicAdd(&deg[bdst[i]], 1.0f);
}

// ---------------- message MLP + scatter-add aggregation ----------------------
__global__ void k_message(const float* __restrict__ x, const float* __restrict__ embl,
                          const int* __restrict__ bt, const int* __restrict__ bsrc,
                          const int* __restrict__ bdst, const __bf16* __restrict__ w1p,
                          const float* __restrict__ b1, const __bf16* __restrict__ w2p,
                          const float* __restrict__ b2, float* __restrict__ aggr, int BE) {
  __shared__ __align__(16) __bf16 sh[16][72];
  int lane = threadIdx.x, m = lane & 15, h = lane >> 4;
  int tile = blockIdx.x;
  int be = tile * 16 + m;
  if (be >= BE) be = BE - 1;
  int di = bdst[be], si = bsrc[be];
  const float* xd = x + (size_t)di * D;
  const float* xs = x + (size_t)si * D;
  const float* tp = embl + (size_t)bt[di] * D;

  v8f acc[4];
#pragma unroll
  for (int nt = 0; nt < 4; nt++) acc[nt] = (v8f){0, 0, 0, 0, 0, 0, 0, 0};
#pragma unroll
  for (int k = 0; k < 6; k++) {   // K = 192: [x_dst | x_src | te_dst]
    int K0 = 32 * k + 8 * h, K1 = K0 + 16;
    const float* p0 = (K0 < 64) ? (xd + K0) : (K0 < 128) ? (xs + (K0 - 64)) : (tp + (K0 - 128));
    const float* p1 = (K1 < 64) ? (xd + K1) : (K1 < 128) ? (xs + (K1 - 64)) : (tp + (K1 - 128));
    v16bf a = gather_a(p0, p1);
#pragma unroll
    for (int nt = 0; nt < 4; nt++)
      acc[nt] = WMMA_BF16(a, load_b(w1p, nt, k, 6, lane), acc[nt]);
  }
#pragma unroll
  for (int nt = 0; nt < 4; nt++) {
    int n = nt * 16 + m;
    float bb = b1[n];
#pragma unroll
    for (int i = 0; i < 8; i++) sh[i + 8 * h][n] = (__bf16)silu_f(acc[nt][i] + bb);
  }
  __syncthreads();

  v8f o[4];
#pragma unroll
  for (int nt = 0; nt < 4; nt++) o[nt] = (v8f){0, 0, 0, 0, 0, 0, 0, 0};
#pragma unroll
  for (int k = 0; k < 2; k++) {
    v16bf a = lds_a(&sh[m][0], k, h);
#pragma unroll
    for (int nt = 0; nt < 4; nt++)
      o[nt] = WMMA_BF16(a, load_b(w2p, nt, k, 2, lane), o[nt]);
  }
  int drow[8];
#pragma unroll
  for (int i = 0; i < 8; i++) {
    int er = tile * 16 + i + 8 * h;
    drow[i] = (er < BE) ? bdst[er] : -1;
  }
#pragma unroll
  for (int nt = 0; nt < 4; nt++) {
    int n = nt * 16 + m;
    float bb = b2[n];
#pragma unroll
    for (int i = 0; i < 8; i++)
      if (drow[i] >= 0) atomicAdd(&aggr[(size_t)drow[i] * D + n], o[nt][i] + bb);
  }
}

// ---------------- update MLP with residual -----------------------------------
__global__ void k_update(float* __restrict__ x, const float* __restrict__ aggr,
                         const float* __restrict__ deg, const __bf16* __restrict__ w1p,
                         const float* __restrict__ b1, const __bf16* __restrict__ w2p,
                         const float* __restrict__ b2, int E) {
  __shared__ __align__(16) __bf16 sh[16][72];
  int lane = threadIdx.x, m = lane & 15, h = lane >> 4;
  int tile = blockIdx.x;
  int e = tile * 16 + m;
  if (e >= E) e = E - 1;
  float inv = 1.0f / fmaxf(deg[e], 1.0f);
  const float* ar = aggr + (size_t)e * D;
  const float* xr = x + (size_t)e * D;

  v8f acc[4];
#pragma unroll
  for (int nt = 0; nt < 4; nt++) acc[nt] = (v8f){0, 0, 0, 0, 0, 0, 0, 0};
#pragma unroll
  for (int k = 0; k < 4; k++) {   // K = 128: [aggr/deg | x]
    int K0 = 32 * k + 8 * h, K1 = K0 + 16;
    const float* p0 = (K0 < 64) ? (ar + K0) : (xr + (K0 - 64));
    const float* p1 = (K1 < 64) ? (ar + K1) : (xr + (K1 - 64));
    float s0 = (K0 < 64) ? inv : 1.0f;
    float s1 = (K1 < 64) ? inv : 1.0f;
    v16bf a = gather_a_scaled(p0, s0, p1, s1);
#pragma unroll
    for (int nt = 0; nt < 4; nt++)
      acc[nt] = WMMA_BF16(a, load_b(w1p, nt, k, 4, lane), acc[nt]);
  }
#pragma unroll
  for (int nt = 0; nt < 4; nt++) {
    int n = nt * 16 + m;
    float bb = b1[n];
#pragma unroll
    for (int i = 0; i < 8; i++) sh[i + 8 * h][n] = (__bf16)silu_f(acc[nt][i] + bb);
  }
  __syncthreads();

  v8f o[4];
#pragma unroll
  for (int nt = 0; nt < 4; nt++) o[nt] = (v8f){0, 0, 0, 0, 0, 0, 0, 0};
#pragma unroll
  for (int k = 0; k < 2; k++) {
    v16bf a = lds_a(&sh[m][0], k, h);
#pragma unroll
    for (int nt = 0; nt < 4; nt++)
      o[nt] = WMMA_BF16(a, load_b(w2p, nt, k, 2, lane), o[nt]);
  }
#pragma unroll
  for (int nt = 0; nt < 4; nt++) {
    int n = nt * 16 + m;
    float bb = b2[n];
#pragma unroll
    for (int i = 0; i < 8; i++) {
      int er = tile * 16 + i + 8 * h;
      if (er < E) {
        size_t off = (size_t)er * D + n;
        x[off] = x[off] + o[nt][i] + bb;
      }
    }
  }
}

// ---------------- graph pooling ----------------------------------------------
__global__ void k_count(const int* __restrict__ erow, const int* __restrict__ batch,
                        float* __restrict__ gcnt, int E, int B) {
  int e = blockIdx.x * blockDim.x + threadIdx.x;
  if (e >= E) return;
  int g = batch[erow[e]];
  if (g >= 0 && g < B) atomicAdd(&gcnt[g], 1.0f);
}

__global__ void k_gaccum(const float* __restrict__ x, const int* __restrict__ erow,
                         const int* __restrict__ batch, float* __restrict__ gsum,
                         int E, int B) {
  int idx = blockIdx.x * blockDim.x + threadIdx.x;
  if (idx >= E * D) return;
  int e = idx >> 6, c = idx & 63;
  int g = batch[erow[e]];
  if (g >= 0 && g < B) atomicAdd(&gsum[g * D + c], x[idx]);
}

__global__ void k_gfinal(const float* __restrict__ gsum, const float* __restrict__ gcnt,
                         float* __restrict__ out, int B) {
  int i = blockIdx.x * blockDim.x + threadIdx.x;
  if (i >= B * D) return;
  out[i] = gsum[i] / fmaxf(gcnt[i >> 6], 1.0f);
}

// -----------------------------------------------------------------------------
extern "C" void kernel_launch(void* const* d_in, const int* in_sizes, int n_in,
                              void* d_out, int out_size, void* d_ws, size_t ws_size,
                              hipStream_t stream) {
  (void)n_in; (void)ws_size;
  const float* af    = (const float*)d_in[0];
  const int*   batch = (const int*)d_in[3];
  const int*   eidx  = (const int*)d_in[4];
  const int*   bei   = (const int*)d_in[5];
  const float* iw1   = (const float*)d_in[6];
  const float* ib1   = (const float*)d_in[7];
  const float* iw2   = (const float*)d_in[8];
  const float* ib2   = (const float*)d_in[9];
  const float* emb   = (const float*)d_in[10];
  const float* mw1   = (const float*)d_in[11];
  const float* mb1   = (const float*)d_in[12];
  const float* mw2   = (const float*)d_in[13];
  const float* mb2   = (const float*)d_in[14];
  const float* uw1   = (const float*)d_in[15];
  const float* ub1   = (const float*)d_in[16];
  const float* uw2   = (const float*)d_in[17];
  const float* ub2   = (const float*)d_in[18];

  int E  = in_sizes[4] / 2;
  int BE = in_sizes[5] / 2;
  int L  = in_sizes[11] / (192 * 64);
  int Bg = (out_size - E * D) / D;

  float* x    = (float*)d_out;
  float* gout = x + (size_t)E * D;

  char* base = (char*)d_ws;
  size_t off = 0;
  auto alloc = [&](size_t bytes) -> char* {
    char* r = base + off;
    off = (off + bytes + 255) & ~(size_t)255;
    return r;
  };
  int*    bt   = (int*)alloc((size_t)E * 4);
  float*  deg  = (float*)alloc((size_t)E * 4);
  float*  aggr = (float*)alloc((size_t)E * D * 4);
  float*  gsum = (float*)alloc((size_t)Bg * D * 4);
  float*  gcnt = (float*)alloc((size_t)Bg * 4);
  __bf16* iw1p = (__bf16*)alloc((size_t)256 * 64 * 2);
  __bf16* iw2p = (__bf16*)alloc((size_t)64 * 64 * 2);
  __bf16* mw1p = (__bf16*)alloc((size_t)L * 192 * 64 * 2);
  __bf16* mw2p = (__bf16*)alloc((size_t)L * 64 * 64 * 2);
  __bf16* uw1p = (__bf16*)alloc((size_t)L * 128 * 64 * 2);
  __bf16* uw2p = (__bf16*)alloc((size_t)L * 64 * 64 * 2);

  hipMemsetAsync(deg, 0, (size_t)E * 4, stream);
  hipMemsetAsync(gsum, 0, (size_t)Bg * D * 4, stream);
  hipMemsetAsync(gcnt, 0, (size_t)Bg * 4, stream);

  auto pack = [&](const float* W, __bf16* O, int K) {
    int tot = K * D;
    k_pack<<<(tot + 255) / 256, 256, 0, stream>>>(W, O, K);
  };
  pack(iw1, iw1p, 256);
  pack(iw2, iw2p, 64);
  for (int l = 0; l < L; l++) {
    pack(mw1 + (size_t)l * 192 * 64, mw1p + (size_t)l * 192 * 64, 192);
    pack(mw2 + (size_t)l * 64 * 64,  mw2p + (size_t)l * 64 * 64,  64);
    pack(uw1 + (size_t)l * 128 * 64, uw1p + (size_t)l * 128 * 64, 128);
    pack(uw2 + (size_t)l * 64 * 64,  uw2p + (size_t)l * 64 * 64,  64);
  }

  k_classify<<<(E + 255) / 256, 256, 0, stream>>>(af, eidx, eidx + E, bt, E);
  k_bond_init<<<(E + 15) / 16, 32, 0, stream>>>(af, eidx, eidx + E, iw1p, ib1, iw2p, ib2, x, E);
  k_deg<<<(BE + 255) / 256, 256, 0, stream>>>(bei + BE, deg, BE);

  for (int l = 0; l < L; l++) {
    hipMemsetAsync(aggr, 0, (size_t)E * D * 4, stream);
    k_message<<<(BE + 15) / 16, 32, 0, stream>>>(
        x, emb + (size_t)l * 5 * D, bt, bei, bei + BE,
        mw1p + (size_t)l * 192 * 64, mb1 + (size_t)l * D,
        mw2p + (size_t)l * 64 * 64,  mb2 + (size_t)l * D, aggr, BE);
    k_update<<<(E + 15) / 16, 32, 0, stream>>>(
        x, aggr, deg,
        uw1p + (size_t)l * 128 * 64, ub1 + (size_t)l * D,
        uw2p + (size_t)l * 64 * 64,  ub2 + (size_t)l * D, E);
  }

  k_count<<<(E + 255) / 256, 256, 0, stream>>>(eidx, batch, gcnt, E, Bg);
  k_gaccum<<<(E * D + 255) / 256, 256, 0, stream>>>(x, eidx, batch, gsum, E, Bg);
  k_gfinal<<<(Bg * D + 255) / 256, 256, 0, stream>>>(gsum, gcnt, gout, Bg);
}
